// NetGATmulti_6777458393523
// MI455X (gfx1250) — compile-verified, matrix operand order checked by reading, not run on hardware
//
#include <hip/hip_runtime.h>
#include <hip/hip_bf16.h>

// ---------------------------------------------------------------------------
// GAT (TransformerConv, 4 heads x 32) + MLP readout for MI455X (gfx1250).
//  K0: init accumulators; convert x->f16; convert+transpose weights->f16
//  K1: fused Q/K/V/skip projections via v_wmma_f32_16x16x32_f16
//      (all fragment halves are single global_load_b128s of contiguous f16)
//  K2: per-edge attention logits: float4 loads, 8-lane-group shfl reduction,
//      float atomic-max (int punning)
//  K3: exp(score - max) + segment-sum denominator (atomicAdd; fits in L2)
//  K4: alpha-weighted scatter of V rows into agg (contiguous f32 atomics,
//      51MB agg buffer resolves inside the 192MB L2)
//  K5: h = tanh(agg + skip); tanhshrink MLP 128->24->8->2, wave-per-node
// ---------------------------------------------------------------------------

typedef __attribute__((ext_vector_type(16))) _Float16 v16h;
typedef __attribute__((ext_vector_type(8)))  _Float16 h8;
typedef __attribute__((ext_vector_type(8)))  float    v8f;

union Frag16 { v16h v; h8 g[2]; _Float16 h[16]; };
union CFrag  { v8f  v; float f[8]; };

#define IN_DIM 128
#define HD_DIM 128
#define HEADS  4
#define HID    32

// ---------------- K0: init + f16 staging -----------------------------------
// t in [0, n*128): agg=0, xh[t]=f16(x[t])
// t in [0, n*4):   denom=0, mmax=-FLT_MAX
// t in [0, 65536): Wt[mat][n*128+k] = f16(W[mat][k*128+n])   (col-major f16)
__global__ __launch_bounds__(256) void gat_init(
    const float* __restrict__ x,
    const float* __restrict__ Wq, const float* __restrict__ Wk,
    const float* __restrict__ Wv, const float* __restrict__ Ws,
    float* __restrict__ agg, float* __restrict__ denom,
    float* __restrict__ mmax, _Float16* __restrict__ xh,
    _Float16* __restrict__ Wt, int n)
{
  int t = blockIdx.x * blockDim.x + threadIdx.x;
  int total = n * HD_DIM;
  if (t < total) { agg[t] = 0.0f; xh[t] = (_Float16)x[t]; }
  if (t < n * HEADS) { denom[t] = 0.0f; mmax[t] = -3.402823466e38f; }
  if (t < 4 * IN_DIM * HD_DIM) {
    int mat = t >> 14;
    int r   = t & 16383;          // r = k*128 + nn
    int kk  = r >> 7;
    int nn  = r & 127;
    const float* Wm = (mat == 0) ? Wq : (mat == 1) ? Wk : (mat == 2) ? Wv : Ws;
    Wt[((size_t)mat << 14) + (size_t)nn * IN_DIM + kk] = (_Float16)Wm[r];
  }
}

// ---------------- K1: fused QKVS GEMM via WMMA -----------------------------
// One wave per 16x16 output tile; K = 128 -> 4 x wmma_f32_16x16x32_f16.
// A fragment (lane, half=lane>>4): halves[0..7]=K[kb0..], halves[8..15]=K[kb1..]
// B fragment identical K striping on column (colBase + lane&15).
__global__ __launch_bounds__(256) void gat_qkvs_gemm(
    const _Float16* __restrict__ xh, const _Float16* __restrict__ Wt,
    const float* __restrict__ bq, const float* __restrict__ bk,
    const float* __restrict__ bv, const float* __restrict__ bs,
    float* __restrict__ qo, float* __restrict__ ko,
    float* __restrict__ vo, float* __restrict__ so, int n)
{
  int wid  = (blockIdx.x * blockDim.x + threadIdx.x) >> 5;
  int lane = threadIdx.x & 31;
  int totalTiles = (n >> 4) * 32;          // (n/16) row tiles * 32 col tiles
  if (wid >= totalTiles) return;

  int tile_n  = wid & 31;                  // 0..31 -> mat(4) x coltile(8)
  int tile_m  = wid >> 5;
  int mat     = tile_n >> 3;
  int colBase = (tile_n & 7) << 4;
  int rowBase = tile_m << 4;

  const float* bias; float* out;
  if      (mat == 0) { bias = bq; out = qo; }
  else if (mat == 1) { bias = bk; out = ko; }
  else if (mat == 2) { bias = bv; out = vo; }
  else               { bias = bs; out = so; }

  int half = lane >> 4;                    // ISA 16-bit A/B K-group select
  int l15  = lane & 15;

  CFrag acc;
  #pragma unroll
  for (int i = 0; i < 8; i++) acc.f[i] = 0.0f;

  const _Float16* xrow = xh + (size_t)(rowBase + l15) * IN_DIM;
  const _Float16* wcol = Wt + ((size_t)mat << 14) + (size_t)(colBase + l15) * IN_DIM;

  #pragma unroll
  for (int kk = 0; kk < IN_DIM; kk += 32) {
    int kb0 = kk + half * 8;               // halves 0..7  -> K[kb0..kb0+7]
    int kb1 = kk + 16 + half * 8;          // halves 8..15 -> K[kb1..kb1+7]

    Frag16 a, b;
    a.g[0] = *(const h8*)(xrow + kb0);     // global_load_b128
    a.g[1] = *(const h8*)(xrow + kb1);
    b.g[0] = *(const h8*)(wcol + kb0);
    b.g[1] = *(const h8*)(wcol + kb1);

    acc.v = __builtin_amdgcn_wmma_f32_16x16x32_f16(
        false, a.v, false, b.v, (short)0, acc.v, false, false);
  }

  float bcol = bias[colBase + l15];
  #pragma unroll
  for (int r = 0; r < 8; r++) {            // D: vgpr r, lanes>=16 -> M+8
    int row = rowBase + half * 8 + r;
    out[(size_t)row * HD_DIM + colBase + l15] = acc.f[r] + bcol;
  }
}

// ---------------- float atomic max via int punning -------------------------
__device__ inline void atomicMaxF32(float* addr, float value) {
  if (value >= 0.0f)
    atomicMax((int*)addr, __float_as_int(value));
  else
    atomicMin((unsigned int*)addr, __float_as_uint(value));
}

// ---------------- K2: edge attention logits + running segment max ----------
// Lane owns contiguous channels 4*lane..4*lane+3 (head = lane>>3): float4
// loads, 3-step xor reduction within each 8-lane head group.
__global__ __launch_bounds__(256) void gat_edge_scores(
    const float* __restrict__ q, const float* __restrict__ k,
    const int* __restrict__ src, const int* __restrict__ dst,
    float* __restrict__ scores, float* __restrict__ mmax, int nE)
{
  int e    = (blockIdx.x * blockDim.x + threadIdx.x) >> 5;
  int lane = threadIdx.x & 31;
  if (e >= nE) return;

  int s = src[e], d = dst[e];
  const float* qd = q + (size_t)d * HD_DIM;
  const float* ks = k + (size_t)s * HD_DIM;

  float4 qv = *(const float4*)(qd + 4 * lane);
  float4 kv = *(const float4*)(ks + 4 * lane);
  float p = qv.x * kv.x + qv.y * kv.y + qv.z * kv.z + qv.w * kv.w;

  #pragma unroll
  for (int off = 4; off > 0; off >>= 1)    // reduce within 8-lane head group
    p += __shfl_xor(p, off, 32);

  float s0 = __shfl(p, 0, 32);
  float s1 = __shfl(p, 8, 32);
  float s2 = __shfl(p, 16, 32);
  float s3 = __shfl(p, 24, 32);

  if (lane == 0) {
    const float scale = 0.17677669529663687f;   // 1/sqrt(32)
    float4 sc = make_float4(s0 * scale, s1 * scale, s2 * scale, s3 * scale);
    *(float4*)(scores + (size_t)e * HEADS) = sc;
    atomicMaxF32(&mmax[d * HEADS + 0], sc.x);
    atomicMaxF32(&mmax[d * HEADS + 1], sc.y);
    atomicMaxF32(&mmax[d * HEADS + 2], sc.z);
    atomicMaxF32(&mmax[d * HEADS + 3], sc.w);
  }
}

// ---------------- K3: exp(score - max), accumulate denominator -------------
__global__ __launch_bounds__(256) void gat_edge_exp(
    float* __restrict__ scores, const float* __restrict__ mmax,
    const int* __restrict__ dst, float* __restrict__ denom, int nE)
{
  int t = blockIdx.x * blockDim.x + threadIdx.x;
  if (t >= nE * HEADS) return;
  int e = t >> 2, h = t & 3;
  int d = dst[e];
  float ev = expf(scores[t] - mmax[d * HEADS + h]);
  scores[t] = ev;
  atomicAdd(&denom[d * HEADS + h], ev);
}

// ---------------- K4: alpha-weighted scatter of V --------------------------
__global__ __launch_bounds__(256) void gat_edge_agg(
    const float* __restrict__ v, const float* __restrict__ scores,
    const float* __restrict__ denom,
    const int* __restrict__ src, const int* __restrict__ dst,
    float* __restrict__ agg, int nE)
{
  int e    = (blockIdx.x * blockDim.x + threadIdx.x) >> 5;
  int lane = threadIdx.x & 31;
  if (e >= nE) return;

  int s = src[e], d = dst[e];
  int h = lane >> 3;                       // head for this lane's channels
  float alpha = scores[(size_t)e * HEADS + h] / denom[d * HEADS + h];

  const float* vs = v   + (size_t)s * HD_DIM + 4 * lane;
  float*       ad = agg + (size_t)d * HD_DIM + 4 * lane;
  float4 vv = *(const float4*)vs;          // global_load_b128

  atomicAdd(&ad[0], alpha * vv.x);         // contiguous; resolves in L2
  atomicAdd(&ad[1], alpha * vv.y);
  atomicAdd(&ad[2], alpha * vv.z);
  atomicAdd(&ad[3], alpha * vv.w);
}

// ---------------- K5: tanh(agg + skip) + tanhshrink MLP readout ------------
__global__ __launch_bounds__(256) void gat_readout(
    const float* __restrict__ agg, const float* __restrict__ skip,
    const float* __restrict__ W1, const float* __restrict__ b1,
    const float* __restrict__ W2, const float* __restrict__ b2,
    const float* __restrict__ W3, const float* __restrict__ b3,
    float* __restrict__ out, int n)
{
  int node = (blockIdx.x * blockDim.x + threadIdx.x) >> 5;
  int lane = threadIdx.x & 31;
  if (node >= n) return;

  const float* a = agg  + (size_t)node * HD_DIM + 4 * lane;
  const float* s = skip + (size_t)node * HD_DIM + 4 * lane;
  float4 av = *(const float4*)a;
  float4 sv = *(const float4*)s;

  float h[4];
  h[0] = tanhf(av.x + sv.x);
  h[1] = tanhf(av.y + sv.y);
  h[2] = tanhf(av.z + sv.z);
  h[3] = tanhf(av.w + sv.w);

  float h1[24];
  #pragma unroll
  for (int j = 0; j < 24; j++) {
    float acc = 0.0f;
    #pragma unroll
    for (int i = 0; i < 4; i++)
      acc += h[i] * W1[(size_t)(4 * lane + i) * 24 + j];
    #pragma unroll
    for (int off = 16; off > 0; off >>= 1)
      acc += __shfl_xor(acc, off, 32);     // all lanes hold full sum
    float z = acc + b1[j];
    h1[j] = z - tanhf(z);                  // tanhshrink
  }

  float h2[8];
  #pragma unroll
  for (int jj = 0; jj < 8; jj++) {
    float acc = b2[jj];
    #pragma unroll
    for (int j = 0; j < 24; j++) acc += h1[j] * W2[j * 8 + jj];
    h2[jj] = acc - tanhf(acc);
  }

  if (lane == 0) {
    #pragma unroll
    for (int oo = 0; oo < 2; oo++) {
      float acc = b3[oo];
      #pragma unroll
      for (int jj = 0; jj < 8; jj++) acc += h2[jj] * W3[jj * 2 + oo];
      out[(size_t)node * 2 + oo] = acc;
    }
  }
}

// ---------------------------------------------------------------------------
extern "C" void kernel_launch(void* const* d_in, const int* in_sizes, int n_in,
                              void* d_out, int out_size, void* d_ws, size_t ws_size,
                              hipStream_t stream) {
  const float* x  = (const float*)d_in[0];
  const int*   ei = (const int*)  d_in[1];
  const float* Wq = (const float*)d_in[2];  const float* bq = (const float*)d_in[3];
  const float* Wk = (const float*)d_in[4];  const float* bk = (const float*)d_in[5];
  const float* Wv = (const float*)d_in[6];  const float* bv = (const float*)d_in[7];
  const float* Ws = (const float*)d_in[8];  const float* bs = (const float*)d_in[9];
  const float* W1 = (const float*)d_in[10]; const float* b1 = (const float*)d_in[11];
  const float* W2 = (const float*)d_in[12]; const float* b2 = (const float*)d_in[13];
  const float* W3 = (const float*)d_in[14]; const float* b3 = (const float*)d_in[15];

  const int n = in_sizes[0] / IN_DIM;       // 100000
  const int e = in_sizes[1] / 2;            // 1600000
  const int* src = ei;                      // edge_index row 0
  const int* dst = ei + e;                  // edge_index row 1

  // workspace layout
  float* ws    = (float*)d_ws;
  size_t N128  = (size_t)n * HD_DIM;
  float* q     = ws;                        // N*128 f32
  float* k     = q + N128;                  // N*128 f32
  float* v     = k + N128;                  // N*128 f32
  float* skp   = v + N128;                  // N*128 f32
  float* agg   = skp + N128;                // N*128 f32
  float* sc    = agg + N128;                // E*4  f32
  float* mmax  = sc + (size_t)e * HEADS;    // N*4  f32
  float* denom = mmax + (size_t)n * HEADS;  // N*4  f32
  _Float16* xh = (_Float16*)(denom + (size_t)n * HEADS);  // N*128 f16 (16B aligned)
  _Float16* Wt = xh + N128;                 // 4*128*128 f16, col-major per mat

  float* out = (float*)d_out;

  const int T = 256;
  // 0) init + f16 staging of x and transposed weights
  {
    int work = n * HD_DIM;
    gat_init<<<(work + T - 1) / T, T, 0, stream>>>(x, Wq, Wk, Wv, Ws,
                                                   agg, denom, mmax, xh, Wt, n);
  }
  // 1) Q/K/V/skip projections (WMMA, b128 fragment loads)
  {
    int tiles = (n >> 4) * 32;              // waves
    int blocks = (tiles + 7) / 8;           // 8 waves / block
    gat_qkvs_gemm<<<blocks, T, 0, stream>>>(xh, Wt, bq, bk, bv, bs,
                                            q, k, v, skp, n);
  }
  // 2) attention logits + segment max (wave per edge)
  {
    int blocks = (e + 7) / 8;
    gat_edge_scores<<<blocks, T, 0, stream>>>(q, k, src, dst, sc, mmax, e);
  }
  // 3) exp + denominator (thread per edge-head)
  {
    int work = e * HEADS;
    gat_edge_exp<<<(work + T - 1) / T, T, 0, stream>>>(sc, mmax, dst, denom, e);
  }
  // 4) weighted scatter aggregate (wave per edge)
  {
    int blocks = (e + 7) / 8;
    gat_edge_agg<<<blocks, T, 0, stream>>>(v, sc, denom, src, dst, agg, e);
  }
  // 5) activation + readout MLP (wave per node)
  {
    int blocks = (n + 7) / 8;
    gat_readout<<<blocks, T, 0, stream>>>(agg, skp, W1, b1, W2, b2, W3, b3,
                                          out, n);
  }
}
